// LSTMModel_22600117911929
// MI455X (gfx1250) — compile-verified, matrix operand order checked by reading, not run on hardware
//
#include <hip/hip_runtime.h>
#include <hip/hip_bf16.h>

// ---------------------------------------------------------------------------
// CDNA5 (gfx1250) implementation: all matmuls via v_wmma_f32_16x16x32_f16
// (f16 operands, f32 accumulate). Weights converted to padded f16 per call.
// Requires ws_size >= ~240 MB of scratch.
// ---------------------------------------------------------------------------

typedef _Float16 f16;
typedef __attribute__((ext_vector_type(16))) _Float16 v16h;
typedef __attribute__((ext_vector_type(8)))  _Float16 v8h;
typedef __attribute__((ext_vector_type(8)))  float    v8f;

#define LB __launch_bounds__(32)

// Fragment loader for 16x16x32 WMMA, wave32.
// A operand: lane L (<16): row m0+L, K = {k0..k0+7, k0+16..k0+23}
//            lane L (>=16): row m0+L-16, K = {k0+8..k0+15, k0+24..k0+31}
// B operand: identical with "row" = column index into W (W stored [N,K] row-major).
__device__ __forceinline__ v16h ldfrag(const f16* __restrict__ base, int ld,
                                       int rc0, int k0, int lane) {
  int r  = rc0 + (lane & 15);
  int kb = k0 + ((lane >> 4) << 3);
  const f16* p = base + (size_t)r * (size_t)ld + kb;
  union { v16h v; struct { v8h lo; v8h hi; } s; } u;
  u.s.lo = *(const v8h*)(p);
  u.s.hi = *(const v8h*)(p + 16);
  return u.v;
}

__device__ __forceinline__ v8f wmma16(v16h a, v16h b, v8f c) {
  return __builtin_amdgcn_wmma_f32_16x16x32_f16(
      /*neg_a=*/false, a, /*neg_b=*/false, b,
      /*c_mod=*/(short)0, c, /*reuse_a=*/false, /*reuse_b=*/false);
}

__device__ __forceinline__ float sigm(float x) { return 1.0f / (1.0f + __expf(-x)); }

// ---------------------------------------------------------------------------
// Generic GEMM: C[M,N] = A[M,Kpad] * B[N,Kpad]^T (+bias) (+relu)
// One wave computes one 16x16 tile.  lda = ldb = Kpad.
// ---------------------------------------------------------------------------
__global__ void LB gemm_wmma(const f16* __restrict__ A, const f16* __restrict__ B,
                             const float* __restrict__ bias,
                             float* __restrict__ Cf, f16* __restrict__ Ch,
                             int Kpad, int ldc, int n_limit, int relu) {
  int lane = threadIdx.x;
  int n0 = blockIdx.x << 4;
  int m0 = blockIdx.y << 4;
  v8f acc = {};
  for (int k = 0; k < Kpad; k += 32) {
    v16h a = ldfrag(A, Kpad, m0, k, lane);
    v16h b = ldfrag(B, Kpad, n0, k, lane);
    acc = wmma16(a, b, acc);
  }
  int col = n0 + (lane & 15);
  if (col >= n_limit) return;
  float bv = bias ? bias[col] : 0.0f;
  int rbase = m0 + ((lane >> 4) << 3);
#pragma unroll
  for (int r = 0; r < 8; ++r) {
    float v = acc[r] + bv;
    if (relu) v = fmaxf(v, 0.0f);
    size_t idx = (size_t)(rbase + r) * (size_t)ldc + col;
    if (Cf) Cf[idx] = v;
    if (Ch) Ch[idx] = (f16)v;
  }
}

// ---------------------------------------------------------------------------
// Fused LSTM recurrent step:  z = xz_t + h_in @ Wh^T ; gates ; update c, h.
// Gate order (PyTorch): i, f, g, o at column offsets 0,H,2H,3H of z.
// One wave per 16x16 hidden tile computes all 4 gate tiles (A frag reused).
// ---------------------------------------------------------------------------
__global__ void LB lstm_step(const float* __restrict__ xz,  // [M,4H] incl. bias
                             const f16* __restrict__ Wh,    // [4H,H]
                             const f16* __restrict__ hin,   // [M,H]
                             float* __restrict__ cst,       // [M,H] in/out
                             f16* __restrict__ hout,        // [M,H]
                             f16* __restrict__ seq,         // [M,H] or null
                             int H) {
  int lane = threadIdx.x;
  int n0 = blockIdx.x << 4;
  int m0 = blockIdx.y << 4;
  v8f zi = {}, zf = {}, zg = {}, zo = {};
  for (int k = 0; k < H; k += 32) {
    v16h a  = ldfrag(hin, H, m0, k, lane);
    v16h bi = ldfrag(Wh, H, n0,         k, lane);
    v16h bf = ldfrag(Wh, H, H + n0,     k, lane);
    v16h bg = ldfrag(Wh, H, 2 * H + n0, k, lane);
    v16h bo = ldfrag(Wh, H, 3 * H + n0, k, lane);
    zi = wmma16(a, bi, zi);
    zf = wmma16(a, bf, zf);
    zg = wmma16(a, bg, zg);
    zo = wmma16(a, bo, zo);
  }
  int col = n0 + (lane & 15);
  int rbase = m0 + ((lane >> 4) << 3);
#pragma unroll
  for (int r = 0; r < 8; ++r) {
    int row = rbase + r;
    const float* x = xz + (size_t)row * (size_t)(4 * H);
    float gi = sigm(zi[r] + x[col]);
    float gf = sigm(zf[r] + x[H + col]);
    float gg = tanhf(zg[r] + x[2 * H + col]);
    float go = sigm(zo[r] + x[3 * H + col]);
    size_t idx = (size_t)row * (size_t)H + col;
    float c = gf * cst[idx] + gi * gg;
    cst[idx] = c;
    float h = go * tanhf(c);
    hout[idx] = (f16)h;
    if (seq) seq[idx] = (f16)h;
  }
}

// ---------------------------------------------------------------------------
// Elementwise / gather helpers
// ---------------------------------------------------------------------------
__global__ void k_fill_f16(f16* p, int n) {
  int i = blockIdx.x * blockDim.x + threadIdx.x;
  if (i < n) p[i] = (f16)0.0f;
}
__global__ void k_fill_f32(float* p, int n) {
  int i = blockIdx.x * blockDim.x + threadIdx.x;
  if (i < n) p[i] = 0.0f;
}
__global__ void k_convert_pad(const float* __restrict__ s, f16* __restrict__ d,
                              int rows, int cols, int rpad, int cpad) {
  int i = blockIdx.x * blockDim.x + threadIdx.x;
  int tot = rpad * cpad;
  if (i >= tot) return;
  int r = i / cpad, c = i % cpad;
  float v = (r < rows && c < cols) ? s[(size_t)r * cols + c] : 0.0f;
  d[i] = (f16)v;
}
// x[t*B+b][0:448] = concat(emb[word], delta_emb[delta], tag_emb[tag], pad)
__global__ void k_build_wordx(const int* wi, const int* di, const int* ti,
                              const float* emb, const float* demb, const float* temb,
                              f16* x) {
  int i = blockIdx.x * blockDim.x + threadIdx.x;
  const int tot = 8192 * 448;
  if (i >= tot) return;
  int r = i / 448, c = i % 448;
  int t = r >> 5, b = r & 31;  // row = t*32 + b
  float v = 0.0f;
  if (c < 300)      v = emb [(size_t)wi[b * 256 + t] * 300 + c];
  else if (c < 400) v = demb[(size_t)di[b * 256 + t] * 100 + (c - 300)];
  else if (c < 420) v = temb[(size_t)ti[b * 256 + t] * 20  + (c - 400)];
  x[i] = (f16)v;
}
// w2c input [8192,960] = concat(word_lstm_out[512], x[420], pad)
__global__ void k_build_w2cin(const f16* wseq, const f16* x, f16* out) {
  int i = blockIdx.x * blockDim.x + threadIdx.x;
  const int tot = 8192 * 960;
  if (i >= tot) return;
  int r = i / 960, c = i % 960;
  f16 v = (f16)0.0f;
  if (c < 512)      v = wseq[(size_t)r * 512 + c];
  else if (c < 932) v = x[(size_t)r * 448 + (c - 512)];
  out[i] = v;
}
// fwd/bwd sequence [5][2048][512]; t=0 -> open_emb, t>=1 -> child k (reversed for bwd)
__global__ void k_build_fbin(const f16* w2co, const int* open_idx, const float* oemb,
                             int rev, f16* out) {
  int i = blockIdx.x * blockDim.x + threadIdx.x;
  const int tot = 5 * 2048 * 512;
  if (i >= tot) return;
  int t = i / (2048 * 512);
  int rem = i - t * 2048 * 512;
  int row = rem / 512, c = rem % 512;
  int b = row >> 6, j = row & 63;  // row = b*64 + j
  f16 v;
  if (t == 0) {
    v = (f16)oemb[(size_t)open_idx[b * 64 + j] * 512 + c];
  } else {
    int k = rev ? (4 - t) : (t - 1);
    int sr = (j * 4 + k) * 32 + b;  // w2c rows are time-major [s][b]
    v = w2co[(size_t)sr * 512 + c];
  }
  out[i] = v;
}
__global__ void k_build_redin(const f16* fh, const f16* bh, f16* out) {
  int i = blockIdx.x * blockDim.x + threadIdx.x;
  const int tot = 2048 * 1024;
  if (i >= tot) return;
  int r = i / 1024, c = i % 1024;
  out[i] = (c < 512) ? fh[(size_t)r * 512 + c] : bh[(size_t)r * 512 + (c - 512)];
}
// hx rows are (b*64+j); cons input is time-major rows (j*32+b)
__global__ void k_reorder_cons(const f16* hx, f16* out) {
  int i = blockIdx.x * blockDim.x + threadIdx.x;
  const int tot = 2048 * 512;
  if (i >= tot) return;
  int r = i / 512, c = i % 512;
  int j = r >> 5, b = r & 31;
  out[i] = hx[(size_t)(b * 64 + j) * 512 + c];
}
__global__ void k_build_transin(const int* tidx, const float* tremb, f16* out) {
  int i = blockIdx.x * blockDim.x + threadIdx.x;
  const int tot = 16384 * 32;
  if (i >= tot) return;
  int r = i / 32, c = i % 32;
  int t = r >> 5, b = r & 31;
  float v = (c < 20) ? tremb[(size_t)tidx[b * 512 + t] * 20 + c] : 0.0f;
  out[i] = (f16)v;
}
// pred_in rows r = b*512 + t ; cols = [word_top | cons_top | tout]
__global__ void k_build_predin(const f16* w2co, const f16* ctop, const f16* tout,
                               f16* out) {
  int i = blockIdx.x * blockDim.x + threadIdx.x;
  const int tot = 16384 * 1152;
  if (i >= tot) return;
  int r = i / 1152, c = i % 1152;
  int b = r / 512, t = r % 512;
  f16 v;
  if (c < 512)       v = w2co[(size_t)(8160 + b) * 512 + c];       // s = 255
  else if (c < 1024) v = ctop[(size_t)b * 512 + (c - 512)];
  else               v = tout[(size_t)(t * 32 + b) * 128 + (c - 1024)];
  out[i] = v;
}

// ---------------------------------------------------------------------------
extern "C" void kernel_launch(void* const* d_in, const int* in_sizes, int n_in,
                              void* d_out, int out_size, void* d_ws, size_t ws_size,
                              hipStream_t stream) {
  (void)in_sizes; (void)n_in; (void)out_size; (void)ws_size;
  auto Ip = [&](int i) { return (const int*)d_in[i]; };
  auto Fp = [&](int i) { return (const float*)d_in[i]; };

  const int *word_idx = Ip(0), *delta_idx = Ip(1), *tag_idx = Ip(2);
  const int *trans_idx = Ip(3), *open_idx = Ip(4);
  const float *emb = Fp(5), *demb = Fp(6), *temb = Fp(7), *tremb = Fp(8), *oemb = Fp(9);
  // LSTM params: (Wi0, Wh0, b0, Wi1, Wh1, b1) per stack
  const float *wWi0 = Fp(10), *wWh0 = Fp(11), *wb0 = Fp(12),
              *wWi1 = Fp(13), *wWh1 = Fp(14), *wb1 = Fp(15);
  const float *tWi0 = Fp(16), *tWh0 = Fp(17), *tb0 = Fp(18),
              *tWi1 = Fp(19), *tWh1 = Fp(20), *tb1 = Fp(21);
  const float *cWi0 = Fp(22), *cWh0 = Fp(23), *cb0 = Fp(24),
              *cWi1 = Fp(25), *cWh1 = Fp(26), *cb1 = Fp(27);
  const float *fWi0 = Fp(28), *fWh0 = Fp(29), *fb0 = Fp(30),
              *fWi1 = Fp(31), *fWh1 = Fp(32), *fb1 = Fp(33);
  const float *bWi0 = Fp(34), *bWh0 = Fp(35), *bb0 = Fp(36),
              *bWi1 = Fp(37), *bWh1 = Fp(38), *bb1 = Fp(39);
  const float *w2cW = Fp(40), *w2cb = Fp(41);
  const float *redW = Fp(42), *redb = Fp(43);
  const float *WW = Fp(44), *Wb = Fp(45);

  // ---- bump allocator over d_ws ----
  char* wp = (char*)d_ws;
  auto allocH = [&](size_t n) { f16* p = (f16*)wp;
    wp += ((n * sizeof(f16)) + 255) & ~(size_t)255; return p; };
  auto allocF = [&](size_t n) { float* p = (float*)wp;
    wp += ((n * sizeof(float)) + 255) & ~(size_t)255; return p; };

  // f16 weight copies (K padded to mult. of 32)
  f16 *wWi0f = allocH(2048 * 448), *wWh0f = allocH(2048 * 512);
  f16 *wWi1f = allocH(2048 * 512), *wWh1f = allocH(2048 * 512);
  f16 *tWi0f = allocH(512 * 32),   *tWh0f = allocH(512 * 128);
  f16 *tWi1f = allocH(512 * 128),  *tWh1f = allocH(512 * 128);
  f16 *cWi0f = allocH(2048 * 512), *cWh0f = allocH(2048 * 512);
  f16 *cWi1f = allocH(2048 * 512), *cWh1f = allocH(2048 * 512);
  f16 *fWi0f = allocH(2048 * 512), *fWh0f = allocH(2048 * 512);
  f16 *fWi1f = allocH(2048 * 512), *fWh1f = allocH(2048 * 512);
  f16 *bWi0f = allocH(2048 * 512), *bWh0f = allocH(2048 * 512);
  f16 *bWi1f = allocH(2048 * 512), *bWh1f = allocH(2048 * 512);
  f16 *w2cWf = allocH(512 * 960), *redWf = allocH(512 * 1024);
  f16 *WWf = allocH(112 * 1152);
  // activations
  f16 *xf     = allocH((size_t)8192 * 448);
  f16 *seqA   = allocH((size_t)10240 * 512);   // layer-0 seq outputs (reused)
  f16 *seqB   = allocH((size_t)8192 * 512);    // word layer-1 out, later tout
  f16 *w2co   = allocH((size_t)8192 * 512);
  f16 *w2cin  = allocH((size_t)8192 * 960);
  f16 *fbin   = allocH((size_t)5 * 2048 * 512);
  f16 *redin  = allocH((size_t)2048 * 1024);
  f16 *hxbuf  = allocH((size_t)2048 * 512);
  f16 *consin = allocH((size_t)2048 * 512);
  f16 *transin= allocH((size_t)16384 * 32);
  f16 *predin = allocH((size_t)16384 * 1152);
  f16 *hA = allocH((size_t)2048 * 512), *hB = allocH((size_t)2048 * 512);
  f16 *fh = allocH((size_t)2048 * 512), *bh = allocH((size_t)2048 * 512);
  f16 *ctop = allocH((size_t)32 * 512);
  float *cbuf  = allocF((size_t)2048 * 512);
  float *xzBig = allocF((size_t)10240 * 2048);

  auto cdiv = [](int a, int b) { return (a + b - 1) / b; };
  auto conv = [&](const float* s, f16* d, int rows, int cols, int rpad, int cpad) {
    int tot = rpad * cpad;
    k_convert_pad<<<cdiv(tot, 256), 256, 0, stream>>>(s, d, rows, cols, rpad, cpad);
  };
  auto gemm = [&](const f16* A, const f16* Bm, const float* bias, float* Cf, f16* Ch,
                  int M, int Npad, int Kpad, int ldc, int nlim, int relu) {
    gemm_wmma<<<dim3(Npad / 16, M / 16), 32, 0, stream>>>(A, Bm, bias, Cf, Ch,
                                                          Kpad, ldc, nlim, relu);
  };
  // Recurrent loop: ping-pong h buffers; optional seq capture / final-h capture.
  auto run_layer = [&](const float* xz, const f16* Wh, int T, int M, int H,
                       f16* seq, f16* final_h) {
    int n = M * H;
    k_fill_f16<<<cdiv(n, 256), 256, 0, stream>>>(hA, n);
    k_fill_f32<<<cdiv(n, 256), 256, 0, stream>>>(cbuf, n);
    f16 *hin = hA, *hout = hB;
    for (int t = 0; t < T; ++t) {
      f16* ho = (final_h && t == T - 1) ? final_h : hout;
      f16* so = seq ? seq + (size_t)t * n : nullptr;
      lstm_step<<<dim3(H / 16, M / 16), 32, 0, stream>>>(
          xz + (size_t)t * M * 4 * H, Wh, hin, cbuf, ho, so, H);
      f16* tmp = hin; hin = ho; hout = tmp;
    }
  };

  // ---- weight conversion (every call; deterministic) ----
  conv(wWi0, wWi0f, 2048, 420, 2048, 448);
  conv(wWh0, wWh0f, 2048, 512, 2048, 512);
  conv(wWi1, wWi1f, 2048, 512, 2048, 512);
  conv(wWh1, wWh1f, 2048, 512, 2048, 512);
  conv(tWi0, tWi0f, 512, 20, 512, 32);
  conv(tWh0, tWh0f, 512, 128, 512, 128);
  conv(tWi1, tWi1f, 512, 128, 512, 128);
  conv(tWh1, tWh1f, 512, 128, 512, 128);
  conv(cWi0, cWi0f, 2048, 512, 2048, 512);
  conv(cWh0, cWh0f, 2048, 512, 2048, 512);
  conv(cWi1, cWi1f, 2048, 512, 2048, 512);
  conv(cWh1, cWh1f, 2048, 512, 2048, 512);
  conv(fWi0, fWi0f, 2048, 512, 2048, 512);
  conv(fWh0, fWh0f, 2048, 512, 2048, 512);
  conv(fWi1, fWi1f, 2048, 512, 2048, 512);
  conv(fWh1, fWh1f, 2048, 512, 2048, 512);
  conv(bWi0, bWi0f, 2048, 512, 2048, 512);
  conv(bWh0, bWh0f, 2048, 512, 2048, 512);
  conv(bWi1, bWi1f, 2048, 512, 2048, 512);
  conv(bWh1, bWh1f, 2048, 512, 2048, 512);
  conv(w2cW, w2cWf, 512, 932, 512, 960);
  conv(redW, redWf, 512, 1024, 512, 1024);
  conv(WW,   WWf,   100, 1152, 112, 1152);

  // ---- word LSTM (2 layers, T=256, batch 32) ----
  k_build_wordx<<<cdiv(8192 * 448, 256), 256, 0, stream>>>(
      word_idx, delta_idx, tag_idx, emb, demb, temb, xf);
  gemm(xf, wWi0f, wb0, xzBig, nullptr, 8192, 2048, 448, 2048, 2048, 0);
  run_layer(xzBig, wWh0f, 256, 32, 512, seqA, nullptr);
  gemm(seqA, wWi1f, wb1, xzBig, nullptr, 8192, 2048, 512, 2048, 2048, 0);
  run_layer(xzBig, wWh1f, 256, 32, 512, seqB, nullptr);
  // w2c projection
  k_build_w2cin<<<cdiv(8192 * 960, 256), 256, 0, stream>>>(seqB, xf, w2cin);
  gemm(w2cin, w2cWf, w2cb, nullptr, w2co, 8192, 512, 960, 512, 512, 0);

  // ---- fwd tree LSTM (2 layers, T=5, batch 2048) ----
  k_build_fbin<<<cdiv(5 * 2048 * 512, 256), 256, 0, stream>>>(w2co, open_idx, oemb, 0, fbin);
  gemm(fbin, fWi0f, fb0, xzBig, nullptr, 10240, 2048, 512, 2048, 2048, 0);
  run_layer(xzBig, fWh0f, 5, 2048, 512, seqA, nullptr);
  gemm(seqA, fWi1f, fb1, xzBig, nullptr, 10240, 2048, 512, 2048, 2048, 0);
  run_layer(xzBig, fWh1f, 5, 2048, 512, nullptr, fh);

  // ---- bwd tree LSTM ----
  k_build_fbin<<<cdiv(5 * 2048 * 512, 256), 256, 0, stream>>>(w2co, open_idx, oemb, 1, fbin);
  gemm(fbin, bWi0f, bb0, xzBig, nullptr, 10240, 2048, 512, 2048, 2048, 0);
  run_layer(xzBig, bWh0f, 5, 2048, 512, seqA, nullptr);
  gemm(seqA, bWi1f, bb1, xzBig, nullptr, 10240, 2048, 512, 2048, 2048, 0);
  run_layer(xzBig, bWh1f, 5, 2048, 512, nullptr, bh);

  // ---- reduce + cons LSTM (2 layers, T=64, batch 32) ----
  k_build_redin<<<cdiv(2048 * 1024, 256), 256, 0, stream>>>(fh, bh, redin);
  gemm(redin, redWf, redb, nullptr, hxbuf, 2048, 512, 1024, 512, 512, 1);
  k_reorder_cons<<<cdiv(2048 * 512, 256), 256, 0, stream>>>(hxbuf, consin);
  gemm(consin, cWi0f, cb0, xzBig, nullptr, 2048, 2048, 512, 2048, 2048, 0);
  run_layer(xzBig, cWh0f, 64, 32, 512, seqA, nullptr);
  gemm(seqA, cWi1f, cb1, xzBig, nullptr, 2048, 2048, 512, 2048, 2048, 0);
  run_layer(xzBig, cWh1f, 64, 32, 512, nullptr, ctop);

  // ---- trans LSTM (2 layers, T=512, batch 32, H=128) ----
  k_build_transin<<<cdiv(16384 * 32, 256), 256, 0, stream>>>(trans_idx, tremb, transin);
  gemm(transin, tWi0f, tb0, xzBig, nullptr, 16384, 512, 32, 512, 512, 0);
  run_layer(xzBig, tWh0f, 512, 32, 128, seqA, nullptr);
  gemm(seqA, tWi1f, tb1, xzBig, nullptr, 16384, 512, 128, 512, 512, 0);
  run_layer(xzBig, tWh1f, 512, 32, 128, seqB /*tout*/, nullptr);

  // ---- final logits ----
  k_build_predin<<<cdiv(16384 * 1152, 256), 256, 0, stream>>>(w2co, ctop, seqB, predin);
  gemm(predin, WWf, Wb, (float*)d_out, nullptr, 16384, 112, 1152, 100, 100, 0);
}
